// Deform_44195213476584
// MI455X (gfx1250) — matrix operand order repaired and checked by reading, not compiled, not run on hardware
//
#include <hip/hip_runtime.h>
#include <hip/hip_bf16.h>

// ---------------- problem dims ----------------
#define B_    16
#define CIN_  64
#define COUT_ 64
#define G_    8
#define H_    130
#define W_    130
#define HO_   128
#define WO_   128
#define OFFC_ 144          // G*2*3*3
#define OFFC_PAD_ 160      // padded to 10 16-row tiles
#define KRED_ 576          // CIN*9
#define NGT_  72           // G * 9 taps
#define NPARM_ (NGT_ * 64) // descriptors per block (4608)

typedef __attribute__((ext_vector_type(16))) _Float16 v16h;
typedef __attribute__((ext_vector_type(8)))  _Float16 half8;
typedef __attribute__((ext_vector_type(8)))  float    v8f;

union H2U { _Float16 h[2]; unsigned u; };

// 16 contiguous f16 -> B fragment (K = h*16 + e)
__device__ __forceinline__ v16h ld_frag16(const _Float16* p) {
    const half8* q = (const half8*)p;
    half8 lo = q[0];
    half8 hi = q[1];
    return __builtin_shufflevector(lo, hi, 0,1,2,3,4,5,6,7,8,9,10,11,12,13,14,15);
}

// A fragment: elements 0..7 at p0, 8..15 at p1 (callers pass p1 = p0 + 8, so K = h*16 + e)
__device__ __forceinline__ v16h ld_frag_split(const _Float16* p0, const _Float16* p1) {
    half8 lo = *(const half8*)p0;
    half8 hi = *(const half8*)p1;
    return __builtin_shufflevector(lo, hi, 0,1,2,3,4,5,6,7,8,9,10,11,12,13,14,15);
}

__device__ __forceinline__ v8f wmma_f16(v16h a, v16h b, v8f c) {
    return __builtin_amdgcn_wmma_f32_16x16x32_f16(false, a, false, b, (short)0, c,
                                                  false, false);
}

// ---------------- weight pre-conversion: f32 -> f16, row-major [M][K] --------
__global__ __launch_bounds__(256) void k_prep(const float* __restrict__ w1,
                                              const float* __restrict__ w2,
                                              const float* __restrict__ wd,
                                              _Float16* __restrict__ A1,
                                              _Float16* __restrict__ A2,
                                              _Float16* __restrict__ A3) {
    int t = blockIdx.x * 256 + threadIdx.x;
    if (t < COUT_ * KRED_) {
        A1[t] = (_Float16)w1[t];   // w1 flat [cout][cin][ky][kx] == [m][k]
        A3[t] = (_Float16)wd[t];   // wd flat [cout][cin][i][j]  == [m][k]
    }
    if (t < OFFC_PAD_ * CIN_) {
        int m = t >> 6;
        A2[t] = (m < OFFC_) ? (_Float16)w2[t] : (_Float16)0.0f;  // w2 flat [144][64]
    }
}

// ---------------- fused conv1 (3x3) + conv2 (1x1 -> offsets) ----------------
// grid: 16 batches * 128 rows * 2 pixel-tiles(64) = 4096 blocks of 256 threads
__global__ __launch_bounds__(256) void k_conv_off(const float* __restrict__ x,
                                                  const float* __restrict__ b1,
                                                  const float* __restrict__ b2,
                                                  const _Float16* __restrict__ A1,
                                                  const _Float16* __restrict__ A2,
                                                  float* __restrict__ off_out) {
    __shared__ __align__(16) _Float16 Bt[64 * 40];  // im2col tile [px][k], padded
    __shared__ __align__(16) _Float16 Yt[64 * 72];  // y1 tile [px][cin], padded

    const int bi  = blockIdx.x;
    const int oxb = (bi & 1) * 64;
    const int oy  = (bi >> 1) & 127;
    const int b   = bi >> 8;

    const int t    = threadIdx.x;
    const int w    = t >> 5;
    const int lane = t & 31;
    const int hw   = lane >> 4;
    const int lm   = lane & 15;

    const int mtile = (w & 3) * 16;      // cout rows of this wave's C tiles
    const int pxb   = (w >> 2) * 32;     // pixel offset (two 16-px subtiles)

    const float* xb = x + (size_t)b * CIN_ * H_ * W_;

    v8f acc0 = {};
    v8f acc1 = {};

    for (int kk = 0; kk < KRED_; kk += 32) {
        __syncthreads();
        // build im2col B tile: 64 px * 32 k, 8 elements/thread, px-coalesced
        #pragma unroll
        for (int it = 0; it < 8; ++it) {
            int idx = it * 256 + t;
            int kl  = idx >> 6;
            int px  = idx & 63;
            int k   = kk + kl;
            int cin = k / 9;
            int tap = k - cin * 9;
            int ti  = tap / 3;
            int tj  = tap - ti * 3;
            float v = xb[(cin * H_ + (oy + ti)) * W_ + (oxb + px + tj)];
            Bt[px * 40 + kl] = (_Float16)v;
        }
        __syncthreads();

        const _Float16* arow = A1 + (mtile + lm) * KRED_ + kk;
        __builtin_prefetch(arow + 32, 0, 1);  // global_prefetch_b8 next K chunk
        v16h a   = ld_frag_split(arow + hw * 16, arow + hw * 16 + 8);
        v16h bf0 = ld_frag16(&Bt[(pxb + lm) * 40 + hw * 16]);
        v16h bf1 = ld_frag16(&Bt[(pxb + 16 + lm) * 40 + hw * 16]);
        acc0 = wmma_f16(a, bf0, acc0);   // A reused for both B fragments
        acc1 = wmma_f16(a, bf1, acc1);
    }

    __syncthreads();
    // deposit y1 tile (+bias) into LDS as [px][cin] f16 -- conv2 B layout
    #pragma unroll
    for (int r = 0; r < 8; ++r) {
        int cout   = mtile + hw * 8 + r;
        float bias = b1[cout];
        Yt[(pxb + lm)      * 72 + cout] = (_Float16)(acc0[r] + bias);
        Yt[(pxb + 16 + lm) * 72 + cout] = (_Float16)(acc1[r] + bias);
    }
    __syncthreads();

    // ---- conv2: off[m, px] = W2[m, c] * y1[c, px] + b2[m]; K = 64 ----
    const int pxb2 = (w & 3) * 16;       // each wave owns one 16-px tile
    const int mh   = (w >> 2) * 5;       // and 5 of the 10 padded M tiles
    v16h yb0 = ld_frag16(&Yt[(pxb2 + lm) * 72 + 0  + hw * 16]);  // K chunk 0
    v16h yb1 = ld_frag16(&Yt[(pxb2 + lm) * 72 + 32 + hw * 16]);  // K chunk 1
    const int px = oxb + pxb2 + lm;

    #pragma unroll
    for (int s = 0; s < 5; ++s) {
        int mb = (mh + s) * 16;
        const _Float16* a2 = A2 + (mb + lm) * CIN_;
        v16h a0 = ld_frag_split(a2 + 0  + hw * 16, a2 + 8  + hw * 16);
        v16h a1 = ld_frag_split(a2 + 32 + hw * 16, a2 + 40 + hw * 16);
        v8f acc = {};
        acc = wmma_f16(a0, yb0, acc);
        acc = wmma_f16(a1, yb1, acc);
        #pragma unroll
        for (int r = 0; r < 8; ++r) {
            int m = mb + hw * 8 + r;
            if (m < OFFC_) {
                off_out[(((size_t)b * OFFC_ + m) * HO_ + oy) * WO_ + px] =
                    acc[r] + b2[m];
            }
        }
    }
}

// ---------------- deformable conv: bilinear im2col + WMMA GEMM --------------
// Phase 0 computes each (group, tap, pixel) sampling descriptor ONCE (the 8
// channels of a group share it), storing mask-premultiplied f16 weights and
// packed clipped corner coords in LDS. The per-element fill then reduces to
// 2 LDS reads + 4 gathers + 4 FMAs.
__global__ __launch_bounds__(256) void k_deform(const float* __restrict__ x,
                                                const float* __restrict__ off,
                                                const _Float16* __restrict__ A3,
                                                float* __restrict__ y_out) {
    __shared__ __align__(16) _Float16 Bt[64 * 40];
    __shared__ __align__(8)  uint2    Pw[NPARM_];  // {w00,w01},{w10,w11} f16x2
    __shared__ unsigned               Pc[NPARM_];  // y0|x0<<8|y1<<16|x1<<24

    const int bi  = blockIdx.x;
    const int oxb = (bi & 1) * 64;
    const int oy  = (bi >> 1) & 127;
    const int b   = bi >> 8;

    const int t    = threadIdx.x;
    const int w    = t >> 5;
    const int lane = t & 31;
    const int hw   = lane >> 4;
    const int lm   = lane & 15;

    const int mtile = (w & 3) * 16;
    const int pxb   = (w >> 2) * 32;

    const float* xb   = x   + (size_t)b * CIN_ * H_ * W_;
    const float* offb = off + (size_t)b * OFFC_ * HO_ * WO_;

    // ---- phase 0: bilinear descriptors for 72 (g,tap) combos x 64 px ----
    #pragma unroll
    for (int it = 0; it < NPARM_ / 256; ++it) {     // 18 iterations
        int e   = it * 256 + t;
        int pxl = e & 63;
        int gt  = e >> 6;                 // 0..71
        int g   = gt / 9;
        int tap = gt - g * 9;
        int ti  = tap / 3;
        int tj  = tap - ti * 3;

        int ch  = ((g * 3 + ti) * 3 + tj) * 2;
        int pix = oy * WO_ + (oxb + pxl);
        float dy = offb[(size_t)ch * (HO_ * WO_) + pix];
        float dx = offb[(size_t)(ch + 1) * (HO_ * WO_) + pix];

        float py  = dy + (float)(oy + ti);
        float pxf = dx + (float)(oxb + pxl + tj);
        float y0f = floorf(py), x0f = floorf(pxf);
        float ly  = py - y0f,   lx  = pxf - x0f;
        int y0 = (int)y0f, x0 = (int)x0f;
        int y1 = y0 + 1,   x1 = x0 + 1;

        float my0 = (y0 >= 0 && y0 < H_) ? 1.f : 0.f;
        float my1 = (y1 >= 0 && y1 < H_) ? 1.f : 0.f;
        float mx0 = (x0 >= 0 && x0 < W_) ? 1.f : 0.f;
        float mx1 = (x1 >= 0 && x1 < W_) ? 1.f : 0.f;

        H2U p0, p1;
        p0.h[0] = (_Float16)((1.f - ly) * (1.f - lx) * my0 * mx0);
        p0.h[1] = (_Float16)((1.f - ly) * lx         * my0 * mx1);
        p1.h[0] = (_Float16)(ly         * (1.f - lx) * my1 * mx0);
        p1.h[1] = (_Float16)(ly         * lx         * my1 * mx1);

        int yc0 = min(max(y0, 0), H_ - 1);
        int yc1 = min(max(y1, 0), H_ - 1);
        int xc0 = min(max(x0, 0), W_ - 1);
        int xc1 = min(max(x1, 0), W_ - 1);

        Pw[e] = make_uint2(p0.u, p1.u);
        Pc[e] = (unsigned)yc0 | ((unsigned)xc0 << 8) |
                ((unsigned)yc1 << 16) | ((unsigned)xc1 << 24);
    }
    __syncthreads();

    v8f acc0 = {};
    v8f acc1 = {};

    for (int kk = 0; kk < KRED_; kk += 32) {
        if (kk) __syncthreads();      // protect Bt reuse (phase-0 sync covers kk=0)
        #pragma unroll
        for (int it = 0; it < 8; ++it) {
            int idx = it * 256 + t;
            int kl  = idx >> 6;
            int pxl = idx & 63;
            int k   = kk + kl;
            int cin = k / 9;
            int tap = k - cin * 9;
            int g   = cin >> 3;

            int e = (g * 9 + tap) * 64 + pxl;
            uint2    pw = Pw[e];
            unsigned pc = Pc[e];
            int yc0 = pc & 255, xc0 = (pc >> 8) & 255;
            int yc1 = (pc >> 16) & 255, xc1 = pc >> 24;

            const float* xc = xb + (size_t)cin * H_ * W_;
            H2U a0, a1; a0.u = pw.x; a1.u = pw.y;
            float v = (float)a0.h[0] * xc[yc0 * W_ + xc0]
                    + (float)a0.h[1] * xc[yc0 * W_ + xc1]
                    + (float)a1.h[0] * xc[yc1 * W_ + xc0]
                    + (float)a1.h[1] * xc[yc1 * W_ + xc1];
            Bt[pxl * 40 + kl] = (_Float16)v;
        }
        __syncthreads();

        const _Float16* arow = A3 + (mtile + lm) * KRED_ + kk;
        __builtin_prefetch(arow + 32, 0, 1);
        v16h a   = ld_frag_split(arow + hw * 16, arow + hw * 16 + 8);
        v16h bf0 = ld_frag16(&Bt[(pxb + lm) * 40 + hw * 16]);
        v16h bf1 = ld_frag16(&Bt[(pxb + 16 + lm) * 40 + hw * 16]);
        acc0 = wmma_f16(a, bf0, acc0);
        acc1 = wmma_f16(a, bf1, acc1);
    }

    const int px = oxb + pxb + lm;
    #pragma unroll
    for (int r = 0; r < 8; ++r) {
        int m = mtile + hw * 8 + r;
        y_out[(((size_t)b * COUT_ + m) * HO_ + oy) * WO_ + px]      = acc0[r];
        y_out[(((size_t)b * COUT_ + m) * HO_ + oy) * WO_ + px + 16] = acc1[r];
    }
}

// ---------------- launcher ----------------
extern "C" void kernel_launch(void* const* d_in, const int* in_sizes, int n_in,
                              void* d_out, int out_size, void* d_ws, size_t ws_size,
                              hipStream_t stream) {
    const float* x  = (const float*)d_in[0];
    const float* w1 = (const float*)d_in[1];
    const float* b1 = (const float*)d_in[2];
    const float* w2 = (const float*)d_in[3];
    const float* b2 = (const float*)d_in[4];
    const float* wd = (const float*)d_in[5];

    // workspace: f16 weight matrices, row-major [M][K]
    _Float16* A1 = reinterpret_cast<_Float16*>(d_ws);                 // [64][576]
    _Float16* A3 = A1 + COUT_ * KRED_;                                // [64][576]
    _Float16* A2 = A3 + COUT_ * KRED_;                                // [160][64]

    float* y_out   = (float*)d_out;                                   // [16,64,128,128]
    float* off_out = y_out + (size_t)B_ * COUT_ * HO_ * WO_;          // [16,144,128,128]

    k_prep<<<(COUT_ * KRED_ + 255) / 256, 256, 0, stream>>>(w1, w2, wd, A1, A2, A3);

    const int nblk = B_ * HO_ * (WO_ / 64);   // 4096
    k_conv_off<<<nblk, 256, 0, stream>>>(x, b1, b2, A1, A2, off_out);
    k_deform  <<<nblk, 256, 0, stream>>>(x, off_out, A3, y_out);
}
